// MoETransformerBlock_55121610277150
// MI455X (gfx1250) — matrix operand order repaired and checked by reading, not compiled, not run on hardware
//
#include <hip/hip_runtime.h>
#include <hip/hip_bf16.h>

#define BATCH   2
#define SEQ     2048
#define TOKENS  (BATCH * SEQ)     // 4096
#define DMODEL  1024
#define DFF     2048
#define NHEADS  16
#define NKV     4
#define HDIM    64
#define KVDIM   (NKV * HDIM)      // 256
#define NEXP    8
#define PAIRS   (TOKENS * 2)      // 8192 (token, expert) pairs
#define APITCH  40                // LDS row pitch in shorts (80B: 16B-aligned, bank-spread)

typedef __attribute__((ext_vector_type(16))) __bf16 v16bf;
typedef __attribute__((ext_vector_type(8)))  float  v8f;

union Frag {
    v16bf v;
    uint4 q[2];
};

__device__ __forceinline__ unsigned short f32_to_bf16(float f) {
    unsigned int u = __float_as_uint(f);
    unsigned int r = (u + 0x7FFFu + ((u >> 16) & 1u)) >> 16;  // RNE
    return (unsigned short)r;
}

__device__ __forceinline__ v8f wmma_bf16(v16bf a, v16bf b, v8f c) {
    return __builtin_amdgcn_wmma_f32_16x16x32_bf16(
        /*neg_a=*/false, a, /*neg_b=*/false, b,
        /*c_mod=*/(short)0, c, /*reuse_a=*/false, /*reuse_b=*/false);
}

// Load 16 bf16 values as two 16B chunks (A-frag and B-frag: both contiguous, k = 16h + s)
__device__ __forceinline__ v16bf load_frag16(const unsigned short* p0, const unsigned short* p1) {
    Frag f;
    f.q[0] = *(const uint4*)p0;
    f.q[1] = *(const uint4*)p1;
    return f.v;
}

// CDNA5 async copy: 16B per lane, global -> LDS, tracked by ASYNCcnt.
// VDST = 32-bit LDS byte address, VADDR = 64-bit global address, no SADDR.
__device__ __forceinline__ void async_copy_b128(unsigned lds_off, const unsigned short* gptr) {
    asm volatile("global_load_async_to_lds_b128 %0, %1, off"
                 :: "v"(lds_off), "v"(gptr) : "memory");
}
__device__ __forceinline__ void wait_async_all() {
    asm volatile("s_wait_asynccnt 0x0" ::: "memory");
}

// ---------------------------------------------------------------------------
// fp32 -> bf16 conversion (weights, done once per launch)
// ---------------------------------------------------------------------------
__global__ void cvt_bf16_kernel(const float* __restrict__ in,
                                unsigned short* __restrict__ out, long long n) {
    long long stride = (long long)gridDim.x * blockDim.x;
    for (long long i = (long long)blockIdx.x * blockDim.x + threadIdx.x; i < n; i += stride)
        out[i] = f32_to_bf16(in[i]);
}

// ---------------------------------------------------------------------------
// RMSNorm: one block per token row of DMODEL
// ---------------------------------------------------------------------------
__global__ __launch_bounds__(256) void rmsnorm_kernel(const float* __restrict__ x,
                                                      const float* __restrict__ w,
                                                      unsigned short* __restrict__ out_b,
                                                      float* __restrict__ out_f) {
    const int t = blockIdx.x;
    const float* xr = x + (long long)t * DMODEL;
    __shared__ float red[8];
    float ss = 0.f;
    for (int d = threadIdx.x; d < DMODEL; d += 256) { float v = xr[d]; ss += v * v; }
    for (int m = 16; m >= 1; m >>= 1) ss += __shfl_xor(ss, m, 32);
    if ((threadIdx.x & 31) == 0) red[threadIdx.x >> 5] = ss;
    __syncthreads();
    float tot = 0.f;
    for (int i = 0; i < 8; ++i) tot += red[i];
    const float r = rsqrtf(tot / (float)DMODEL + 1e-6f);
    for (int d = threadIdx.x; d < DMODEL; d += 256) {
        float v = xr[d] * r * w[d];
        out_b[(long long)t * DMODEL + d] = f32_to_bf16(v);
        if (out_f) out_f[(long long)t * DMODEL + d] = v;
    }
}

// ---------------------------------------------------------------------------
// Wave-tiled WMMA GEMM with double-buffered async-to-LDS staging:
//   C[m,n] = sum_k A[row(m),k] * W[z][n,k] (+res)
// Block = 256 thr = 8 waves as 4(M)x2(N); wave tile 32x32; block tile 128x64.
// Per 32-deep K-slab: A tile 128x32 + B tile 64x32 staged in LDS via
// GLOBAL_LOAD_ASYNC_TO_LDS_B128 (3 async ops/thread), double-buffered so the
// next slab's copies overlap the current slab's 4 WMMAs.
// ---------------------------------------------------------------------------
__global__ __launch_bounds__(256) void gemm_bf16_kernel(
    const unsigned short* __restrict__ A,
    const unsigned short* __restrict__ W,
    float* __restrict__ C,
    const float* __restrict__ residual,
    const int* __restrict__ rowlist,
    const int* __restrict__ d_count,
    const int* __restrict__ d_offset,
    int M, int N, int K, long long wz)
{
    __shared__ __attribute__((aligned(16))) unsigned short lsA[2][128 * APITCH];
    __shared__ __attribute__((aligned(16))) unsigned short lsB[2][64 * APITCH];

    const int z = blockIdx.z;
    const int Mz = d_count ? d_count[z] : M;
    const int rowBase = d_offset ? d_offset[z] : 0;
    const int mBlock = blockIdx.y * 128;
    const int nBlock = blockIdx.x * 64;
    if (mBlock >= Mz) return;                    // block-uniform exit (before barriers)
    const unsigned short* Wz = W + (long long)z * wz;
    const int wave = threadIdx.x >> 5;
    const int lane = threadIdx.x & 31;
    const int lm = lane & 15, khf = lane >> 4;
    const int mw = (wave >> 1) * 32;             // wave M offset within block tile
    const int nw = (wave & 1) * 32;              // wave N offset within block tile

    // Per-thread async-copy assignments.
    // A tile: 128 rows x 64B = 512 x 16B chunks -> 2 per thread.
    long long agOff[2];
    int aLds[2];
    for (int c = 0; c < 2; ++c) {
        const int chunk = threadIdx.x + c * 256;
        const int row = chunk >> 2, cc = chunk & 3;
        int mg = mBlock + row;
        if (mg >= Mz) mg = Mz - 1;               // duplicate last row; stores are guarded
        const long long rr = rowlist ? (long long)rowlist[rowBase + mg]
                                     : (long long)(rowBase + mg);
        agOff[c] = rr * K + cc * 8;
        aLds[c]  = (row * APITCH + cc * 8) * 2;  // bytes
    }
    // B tile: 64 rows x 64B = 256 x 16B chunks -> 1 per thread.
    const int brow = threadIdx.x >> 2, bcc = threadIdx.x & 3;
    const long long bgOff = (long long)(nBlock + brow) * K + bcc * 8;
    const int bLds = (brow * APITCH + bcc * 8) * 2;

    unsigned baseA[2], baseB[2];
    for (int i = 0; i < 2; ++i) {
        baseA[i] = (unsigned)(unsigned long long)(const void*)&lsA[i][0];
        baseB[i] = (unsigned)(unsigned long long)(const void*)&lsB[i][0];
    }

    v8f zero = {0.f,0.f,0.f,0.f,0.f,0.f,0.f,0.f};
    v8f acc[2][2] = {{zero, zero}, {zero, zero}};

    const int nk = K / 32;
    // prologue: stage slab 0
    async_copy_b128(baseA[0] + aLds[0], A + agOff[0]);
    async_copy_b128(baseA[0] + aLds[1], A + agOff[1]);
    async_copy_b128(baseB[0] + bLds,    Wz + bgOff);

    for (int kt = 0; kt < nk; ++kt) {
        const int cur = kt & 1;
        wait_async_all();
        __syncthreads();                          // slab[cur] visible to all waves
        if (kt + 1 < nk) {                        // stage slab[kt+1] under the WMMAs
            const int k0 = (kt + 1) * 32;
            async_copy_b128(baseA[cur ^ 1] + aLds[0], A + agOff[0] + k0);
            async_copy_b128(baseA[cur ^ 1] + aLds[1], A + agOff[1] + k0);
            async_copy_b128(baseB[cur ^ 1] + bLds,    Wz + bgOff + k0);
        }
        const unsigned short* As = &lsA[cur][0];
        const unsigned short* Bs = &lsB[cur][0];
        const unsigned short* ar0 = As + (mw + lm) * APITCH;
        const unsigned short* ar1 = As + (mw + 16 + lm) * APITCH;
        const unsigned short* br0 = Bs + (nw + lm) * APITCH + khf * 16;
        const unsigned short* br1 = Bs + (nw + 16 + lm) * APITCH + khf * 16;
        v16bf a0 = load_frag16(ar0 + khf * 16, ar0 + khf * 16 + 8);
        v16bf a1 = load_frag16(ar1 + khf * 16, ar1 + khf * 16 + 8);
        v16bf b0 = load_frag16(br0, br0 + 8);
        v16bf b1 = load_frag16(br1, br1 + 8);
        acc[0][0] = wmma_bf16(a0, b0, acc[0][0]);
        acc[0][1] = wmma_bf16(a0, b1, acc[0][1]);
        acc[1][0] = wmma_bf16(a1, b0, acc[1][0]);
        acc[1][1] = wmma_bf16(a1, b1, acc[1][1]);
        // top-of-loop barrier doubles as the write/read fence for the next slab
    }

    for (int mi = 0; mi < 2; ++mi)
        for (int r = 0; r < 8; ++r) {
            const int mg = mBlock + mw + mi * 16 + r + 8 * khf;
            if (mg >= Mz) continue;
            const long long crow = (long long)(rowBase + mg) * N;
            for (int ni = 0; ni < 2; ++ni) {
                const long long off = crow + nBlock + nw + ni * 16 + lm;
                float val = acc[mi][ni][r];
                if (residual) val += residual[off];
                C[off] = val;
            }
        }
}

// ---------------------------------------------------------------------------
// RoPE + layout: q[t,1024]->qh[b,h,s,64] bf16 (roped); k -> kh[b,kvh,s,64];
// v -> vT[b,kvh,64,s] (transposed for the PV WMMA B-fragment).
// ---------------------------------------------------------------------------
__global__ __launch_bounds__(256) void rope_layout_kernel(
    const float* __restrict__ q, const float* __restrict__ k, const float* __restrict__ v,
    unsigned short* __restrict__ qh, unsigned short* __restrict__ kh,
    unsigned short* __restrict__ vt)
{
    const int t = blockIdx.x;
    const int b = t / SEQ, s = t - b * SEQ;
    for (int p = threadIdx.x; p < NHEADS * 32; p += 256) {
        const int h = p >> 5, j = p & 31;
        const float x1 = q[(long long)t * DMODEL + h * HDIM + 2 * j];
        const float x2 = q[(long long)t * DMODEL + h * HDIM + 2 * j + 1];
        const int k2 = (2 * j) & 31;                         // emb[0::2] pattern
        const float theta = (float)s * powf(10000.0f, -(float)k2 / 32.0f);
        float sn, cs; sincosf(theta, &sn, &cs);
        const long long base = (((long long)b * NHEADS + h) * SEQ + s) * HDIM;
        qh[base + 2 * j]     = f32_to_bf16(x1 * cs - x2 * sn);
        qh[base + 2 * j + 1] = f32_to_bf16(x1 * sn + x2 * cs);
    }
    for (int p = threadIdx.x; p < NKV * 32; p += 256) {
        const int h = p >> 5, j = p & 31;
        const float x1 = k[(long long)t * KVDIM + h * HDIM + 2 * j];
        const float x2 = k[(long long)t * KVDIM + h * HDIM + 2 * j + 1];
        const int k2 = (2 * j) & 31;
        const float theta = (float)s * powf(10000.0f, -(float)k2 / 32.0f);
        float sn, cs; sincosf(theta, &sn, &cs);
        const long long base = (((long long)b * NKV + h) * SEQ + s) * HDIM;
        kh[base + 2 * j]     = f32_to_bf16(x1 * cs - x2 * sn);
        kh[base + 2 * j + 1] = f32_to_bf16(x1 * sn + x2 * cs);
    }
    for (int d = threadIdx.x; d < KVDIM; d += 256) {
        const int h = d >> 6, dl = d & 63;
        vt[(((long long)b * NKV + h) * HDIM + dl) * SEQ + s] =
            f32_to_bf16(v[(long long)t * KVDIM + d]);
    }
}

// ---------------------------------------------------------------------------
// Flash attention: 1 wave per (b, h, 16 q-rows); online softmax; causal mask.
// QK^T: 4 WMMAs per 32-key tile. P goes through LDS to switch C->A layout.
// PV: 4 WMMAs per tile against pre-transposed V.
// ---------------------------------------------------------------------------
__global__ __launch_bounds__(32) void flash_attn_kernel(
    const unsigned short* __restrict__ qh,
    const unsigned short* __restrict__ kh,
    const unsigned short* __restrict__ vt,
    unsigned short* __restrict__ attn_out)   // [B, S, 1024] bf16 (head-concat)
{
    __shared__ __attribute__((aligned(16))) unsigned short Pl[16 * 32];
    const int qt = blockIdx.x, h = blockIdx.y, b = blockIdx.z;
    const int lane = threadIdx.x & 31;
    const int lm = lane & 15, khf = lane >> 4;
    const int kvh = h >> 2;                  // GQA: 4 q heads per kv head
    const int q0 = qt * 16;
    const unsigned short* Q  = qh + (((long long)b * NHEADS + h) * SEQ) * HDIM;
    const unsigned short* Kp = kh + (((long long)b * NKV + kvh) * SEQ) * HDIM;
    const unsigned short* Vt = vt + (((long long)b * NKV + kvh) * HDIM) * SEQ;

    v16bf aq[2];
    {
        const unsigned short* qr = Q + (long long)(q0 + lm) * HDIM;
        aq[0] = load_frag16(qr + khf * 16,      qr + khf * 16 + 8);
        aq[1] = load_frag16(qr + 32 + khf * 16, qr + 32 + khf * 16 + 8);
    }
    v8f zero = {0.f,0.f,0.f,0.f,0.f,0.f,0.f,0.f};
    v8f o[4] = {zero, zero, zero, zero};
    float mrow[8], lrow[8];
    for (int r = 0; r < 8; ++r) { mrow[r] = -3.0e38f; lrow[r] = 0.f; }

    const int ntiles = (q0 + 15) / 32 + 1;
    for (int j = 0; j < ntiles; ++j) {
        const int kb = j * 32;
        v8f s0 = zero, s1 = zero;
        {
            const unsigned short* kr0 = Kp + (long long)(kb + lm) * HDIM;
            const unsigned short* kr1 = Kp + (long long)(kb + 16 + lm) * HDIM;
            v16bf b00 = load_frag16(kr0 + khf * 16,      kr0 + khf * 16 + 8);
            v16bf b01 = load_frag16(kr0 + 32 + khf * 16, kr0 + 32 + khf * 16 + 8);
            v16bf b10 = load_frag16(kr1 + khf * 16,      kr1 + khf * 16 + 8);
            v16bf b11 = load_frag16(kr1 + 32 + khf * 16, kr1 + 32 + khf * 16 + 8);
            s0 = wmma_bf16(aq[0], b00, s0);
            s0 = wmma_bf16(aq[1], b01, s0);
            s1 = wmma_bf16(aq[0], b10, s1);
            s1 = wmma_bf16(aq[1], b11, s1);
        }
        for (int r = 0; r < 8; ++r) {
            const int qrow = q0 + r + 8 * khf;
            float e0 = s0[r] * 0.125f;                 // 1/sqrt(64)
            float e1 = s1[r] * 0.125f;
            if (kb + lm > qrow)      e0 = -3.0e38f;
            if (kb + 16 + lm > qrow) e1 = -3.0e38f;
            float tmax = fmaxf(e0, e1);
            tmax = fmaxf(tmax, __shfl_xor(tmax, 1, 32));
            tmax = fmaxf(tmax, __shfl_xor(tmax, 2, 32));
            tmax = fmaxf(tmax, __shfl_xor(tmax, 4, 32));
            tmax = fmaxf(tmax, __shfl_xor(tmax, 8, 32));
            const float mnew = fmaxf(mrow[r], tmax);
            const float sc = __expf(mrow[r] - mnew);
            const float p0 = __expf(e0 - mnew);
            const float p1 = __expf(e1 - mnew);
            float ps = p0 + p1;
            ps += __shfl_xor(ps, 1, 32);
            ps += __shfl_xor(ps, 2, 32);
            ps += __shfl_xor(ps, 4, 32);
            ps += __shfl_xor(ps, 8, 32);
            lrow[r] = lrow[r] * sc + ps;
            mrow[r] = mnew;
            o[0][r] *= sc; o[1][r] *= sc; o[2][r] *= sc; o[3][r] *= sc;
            const int ml = r + 8 * khf;
            Pl[ml * 32 + lm]      = f32_to_bf16(p0);
            Pl[ml * 32 + 16 + lm] = f32_to_bf16(p1);
        }
        __syncthreads();
        v16bf ap = load_frag16(Pl + lm * 32 + khf * 16, Pl + lm * 32 + khf * 16 + 8);
        for (int t = 0; t < 4; ++t) {
            const unsigned short* vr = Vt + (long long)(t * 16 + lm) * SEQ + kb + khf * 16;
            v16bf bv = load_frag16(vr, vr + 8);
            o[t] = wmma_bf16(ap, bv, o[t]);
        }
        __syncthreads();
    }
    for (int r = 0; r < 8; ++r) {
        const float inv = 1.0f / lrow[r];
        const long long tok = (long long)b * SEQ + q0 + r + 8 * khf;
        for (int t = 0; t < 4; ++t)
            attn_out[tok * DMODEL + h * HDIM + t * 16 + lm] = f32_to_bf16(o[t][r] * inv);
    }
}

// ---------------------------------------------------------------------------
// Router: 1 wave per token; fp32 logits so top-k selection matches reference.
// ---------------------------------------------------------------------------
__global__ __launch_bounds__(32) void router_kernel(
    const float* __restrict__ xn2, const float* __restrict__ rw,
    float* __restrict__ probs, int* __restrict__ tki, float* __restrict__ tkw)
{
    const int t = blockIdx.x;
    const int lane = threadIdx.x;
    float acc[NEXP];
    for (int e = 0; e < NEXP; ++e) acc[e] = 0.f;
    for (int d = lane; d < DMODEL; d += 32) {
        const float xv = xn2[(long long)t * DMODEL + d];
        for (int e = 0; e < NEXP; ++e) acc[e] += xv * rw[e * DMODEL + d];
    }
    for (int e = 0; e < NEXP; ++e)
        for (int m = 16; m >= 1; m >>= 1) acc[e] += __shfl_xor(acc[e], m, 32);
    if (lane == 0) {
        float mx = acc[0];
        for (int e = 1; e < NEXP; ++e) mx = fmaxf(mx, acc[e]);
        float p[NEXP], se = 0.f;
        for (int e = 0; e < NEXP; ++e) { p[e] = expf(acc[e] - mx); se += p[e]; }
        for (int e = 0; e < NEXP; ++e) { p[e] /= se; probs[t * NEXP + e] = p[e]; }
        int i0 = 0;
        for (int e = 1; e < NEXP; ++e) if (p[e] > p[i0]) i0 = e;     // ties: lowest idx
        int i1 = (i0 == 0) ? 1 : 0;
        for (int e = 0; e < NEXP; ++e) if (e != i0 && p[e] > p[i1]) i1 = e;
        const float s2 = p[i0] + p[i1];
        tki[t * 2] = i0; tki[t * 2 + 1] = i1;
        tkw[t * 2] = p[i0] / s2; tkw[t * 2 + 1] = p[i1] / s2;
    }
}

// Deterministic per-expert prob sums (fixed tree reduction, no atomics)
__global__ __launch_bounds__(256) void probsum_kernel(const float* __restrict__ probs,
                                                      float* __restrict__ probsum) {
    const int e = blockIdx.x;
    __shared__ float sm[256];
    float s = 0.f;
    for (int t = threadIdx.x; t < TOKENS; t += 256) s += probs[t * NEXP + e];
    sm[threadIdx.x] = s;
    __syncthreads();
    for (int w = 128; w >= 1; w >>= 1) {
        if (threadIdx.x < w) sm[threadIdx.x] += sm[threadIdx.x + w];
        __syncthreads();
    }
    if (threadIdx.x == 0) probsum[e] = sm[0];
}

// Deterministic serial list build (8192 iters; trivially cheap vs GEMMs)
__global__ void build_lists_kernel(const int* __restrict__ tki,
                                   int* __restrict__ counts, int* __restrict__ offsets,
                                   int* __restrict__ rowlist, int* __restrict__ slotmap) {
    if (threadIdx.x != 0 || blockIdx.x != 0) return;
    int cnt[NEXP];
    for (int e = 0; e < NEXP; ++e) cnt[e] = 0;
    for (int t = 0; t < TOKENS; ++t) { cnt[tki[t * 2]]++; cnt[tki[t * 2 + 1]]++; }
    int off[NEXP], run = 0;
    for (int e = 0; e < NEXP; ++e) { off[e] = run; run += cnt[e]; }
    for (int e = 0; e < NEXP; ++e) { counts[e] = cnt[e]; offsets[e] = off[e]; }
    int cur[NEXP];
    for (int e = 0; e < NEXP; ++e) cur[e] = off[e];
    for (int t = 0; t < TOKENS; ++t)
        for (int kk = 0; kk < 2; ++kk) {
            const int e = tki[t * 2 + kk];
            const int slot = cur[e]++;
            rowlist[slot] = t;
            slotmap[t * 2 + kk] = slot;
        }
}

// ---------------------------------------------------------------------------
// Fused gate/up expert GEMM with SiLU*u epilogue -> h bf16 in slot space.
// Same tiling as gemm_bf16_kernel; 8 WMMAs per K-step (2A x (2Bg + 2Bu)).
// ---------------------------------------------------------------------------
__global__ __launch_bounds__(256) void moe_gu_kernel(
    const unsigned short* __restrict__ A,     // xn2 bf16 [TOKENS, DMODEL]
    const unsigned short* __restrict__ Wgp,   // [E, DFF, DMODEL] bf16
    const unsigned short* __restrict__ Wup,
    unsigned short* __restrict__ H,           // [PAIRS, DFF] bf16
    const int* __restrict__ rowlist,
    const int* __restrict__ counts,
    const int* __restrict__ offsets)
{
    const int N = DFF, K = DMODEL;
    const int z = blockIdx.z;
    const int Mz = counts[z];
    const int rowBase = offsets[z];
    const int wave = threadIdx.x >> 5, lane = threadIdx.x & 31;
    const int m0 = blockIdx.y * 128 + (wave >> 1) * 32;
    const int n0 = blockIdx.x * 64 + (wave & 1) * 32;
    if (m0 >= Mz) return;
    const unsigned short* Wgz = Wgp + (long long)z * N * K;
    const unsigned short* Wuz = Wup + (long long)z * N * K;
    const int lm = lane & 15, khf = lane >> 4;
    long long arow[2], wrow[2];
    for (int mi = 0; mi < 2; ++mi) {
        int mg = m0 + mi * 16 + lm;
        if (mg >= Mz) mg = Mz - 1;
        arow[mi] = (long long)rowlist[rowBase + mg] * K;
    }
    for (int ni = 0; ni < 2; ++ni) wrow[ni] = (long long)(n0 + ni * 16 + lm) * K;
    v8f zero = {0.f,0.f,0.f,0.f,0.f,0.f,0.f,0.f};
    v8f cg[2][2] = {{zero, zero}, {zero, zero}};
    v8f cu[2][2] = {{zero, zero}, {zero, zero}};
#pragma unroll 2
    for (int k0 = 0; k0 < K; k0 += 32) {
        v16bf a0 = load_frag16(A + arow[0] + k0 + khf * 16, A + arow[0] + k0 + khf * 16 + 8);
        v16bf a1 = load_frag16(A + arow[1] + k0 + khf * 16, A + arow[1] + k0 + khf * 16 + 8);
        const unsigned short* g0 = Wgz + wrow[0] + k0 + khf * 16;
        const unsigned short* g1 = Wgz + wrow[1] + k0 + khf * 16;
        const unsigned short* u0 = Wuz + wrow[0] + k0 + khf * 16;
        const unsigned short* u1 = Wuz + wrow[1] + k0 + khf * 16;
        v16bf bg0 = load_frag16(g0, g0 + 8), bg1 = load_frag16(g1, g1 + 8);
        v16bf bu0 = load_frag16(u0, u0 + 8), bu1 = load_frag16(u1, u1 + 8);
        cg[0][0] = wmma_bf16(a0, bg0, cg[0][0]);
        cg[0][1] = wmma_bf16(a0, bg1, cg[0][1]);
        cg[1][0] = wmma_bf16(a1, bg0, cg[1][0]);
        cg[1][1] = wmma_bf16(a1, bg1, cg[1][1]);
        cu[0][0] = wmma_bf16(a0, bu0, cu[0][0]);
        cu[0][1] = wmma_bf16(a0, bu1, cu[0][1]);
        cu[1][0] = wmma_bf16(a1, bu0, cu[1][0]);
        cu[1][1] = wmma_bf16(a1, bu1, cu[1][1]);
    }
    for (int mi = 0; mi < 2; ++mi)
        for (int r = 0; r < 8; ++r) {
            const int mg = m0 + mi * 16 + r + 8 * khf;
            if (mg >= Mz) continue;
            const long long crow = (long long)(rowBase + mg) * N;
            for (int ni = 0; ni < 2; ++ni) {
                const float g = cg[mi][ni][r], u = cu[mi][ni][r];
                const float hv = g / (1.0f + __expf(-g)) * u;   // silu(g)*u
                H[crow + n0 + ni * 16 + lm] = f32_to_bf16(hv);
            }
        }
}

// out = (x + attn) + w0*y[slot0] + w1*y[slot1]
__global__ __launch_bounds__(256) void combine_kernel(
    const float* __restrict__ h1, const float* __restrict__ y,
    const int* __restrict__ slotmap, const float* __restrict__ tkw,
    float* __restrict__ outx)
{
    const int t = blockIdx.x;
    const long long s0 = slotmap[t * 2], s1 = slotmap[t * 2 + 1];
    const float w0 = tkw[t * 2], w1 = tkw[t * 2 + 1];
    for (int d = threadIdx.x; d < DMODEL; d += 256)
        outx[(long long)t * DMODEL + d] = h1[(long long)t * DMODEL + d]
            + w0 * y[s0 * DMODEL + d] + w1 * y[s1 * DMODEL + d];
}

__global__ void finalize_kernel(const int* __restrict__ counts,
                                const float* __restrict__ probsum,
                                float* __restrict__ out_aux, float* __restrict__ out_tpe) {
    if (threadIdx.x != 0 || blockIdx.x != 0) return;
    const float tot = (float)PAIRS;
    float aux = 0.f;
    for (int e = 0; e < NEXP; ++e) {
        const float f = (float)counts[e] / tot;
        const float P = probsum[e] / (float)TOKENS;
        aux += f * P;
        out_tpe[e] = (float)counts[e];
    }
    out_aux[0] = (float)NEXP * aux;
}

// ---------------------------------------------------------------------------
extern "C" void kernel_launch(void* const* d_in, const int* in_sizes, int n_in,
                              void* d_out, int out_size, void* d_ws, size_t ws_size,
                              hipStream_t stream) {
    (void)in_sizes; (void)n_in; (void)out_size; (void)ws_size;
    const float* x   = (const float*)d_in[0];
    const float* wq  = (const float*)d_in[1];
    const float* wk  = (const float*)d_in[2];
    const float* wv  = (const float*)d_in[3];
    const float* wo  = (const float*)d_in[4];
    const float* anw = (const float*)d_in[5];
    const float* fnw = (const float*)d_in[6];
    const float* rw  = (const float*)d_in[7];
    const float* Wg  = (const float*)d_in[8];
    const float* Wu  = (const float*)d_in[9];
    const float* Wd  = (const float*)d_in[10];

    char* ws = (char*)d_ws;
    size_t off = 0;
    auto alloc = [&](size_t bytes) -> void* {
        void* p = ws + off;
        off = (off + bytes + 255) & ~(size_t)255;
        return p;
    };
    unsigned short* wq_b  = (unsigned short*)alloc((size_t)DMODEL * DMODEL * 2);
    unsigned short* wk_b  = (unsigned short*)alloc((size_t)KVDIM * DMODEL * 2);
    unsigned short* wv_b  = (unsigned short*)alloc((size_t)KVDIM * DMODEL * 2);
    unsigned short* wo_b  = (unsigned short*)alloc((size_t)DMODEL * DMODEL * 2);
    unsigned short* Wg_b  = (unsigned short*)alloc((size_t)NEXP * DFF * DMODEL * 2);
    unsigned short* Wu_b  = (unsigned short*)alloc((size_t)NEXP * DFF * DMODEL * 2);
    unsigned short* Wd_b  = (unsigned short*)alloc((size_t)NEXP * DMODEL * DFF * 2);
    unsigned short* xn1_b = (unsigned short*)alloc((size_t)TOKENS * DMODEL * 2);
    float* qf             = (float*)alloc((size_t)TOKENS * DMODEL * 4);
    float* kf             = (float*)alloc((size_t)TOKENS * KVDIM * 4);
    float* vf             = (float*)alloc((size_t)TOKENS * KVDIM * 4);
    unsigned short* qh_b  = (unsigned short*)alloc((size_t)TOKENS * DMODEL * 2);
    unsigned short* kh_b  = (unsigned short*)alloc((size_t)TOKENS * KVDIM * 2);
    unsigned short* vt_b  = (unsigned short*)alloc((size_t)TOKENS * KVDIM * 2);
    unsigned short* attn_b= (unsigned short*)alloc((size_t)TOKENS * DMODEL * 2);
    float* h1             = (float*)alloc((size_t)TOKENS * DMODEL * 4);
    unsigned short* xn2_b = (unsigned short*)alloc((size_t)TOKENS * DMODEL * 2);
    float* xn2_f          = (float*)alloc((size_t)TOKENS * DMODEL * 4);
    float* probs          = (float*)alloc((size_t)TOKENS * NEXP * 4);
    float* probsum        = (float*)alloc((size_t)NEXP * 4);
    int*   tki            = (int*)alloc((size_t)TOKENS * 2 * 4);
    float* tkw            = (float*)alloc((size_t)TOKENS * 2 * 4);
    int*   counts         = (int*)alloc((size_t)NEXP * 4);
    int*   offsets        = (int*)alloc((size_t)NEXP * 4);
    int*   rowlist        = (int*)alloc((size_t)PAIRS * 4);
    int*   slotmap        = (int*)alloc((size_t)TOKENS * 2 * 4);
    unsigned short* hbuf  = (unsigned short*)alloc((size_t)PAIRS * DFF * 2);
    float* ybuf           = (float*)alloc((size_t)PAIRS * DMODEL * 4);

    float* out_x   = (float*)d_out;
    float* out_aux = out_x + (size_t)TOKENS * DMODEL;
    float* out_tpe = out_aux + 1;

    auto cvt = [&](const float* in, unsigned short* out, long long n) {
        long long blocks = (n + 1023) / 1024;
        if (blocks > 16384) blocks = 16384;
        cvt_bf16_kernel<<<(int)blocks, 256, 0, stream>>>(in, out, n);
    };
    cvt(wq, wq_b, (long long)DMODEL * DMODEL);
    cvt(wk, wk_b, (long long)KVDIM * DMODEL);
    cvt(wv, wv_b, (long long)KVDIM * DMODEL);
    cvt(wo, wo_b, (long long)DMODEL * DMODEL);
    cvt(Wg, Wg_b, (long long)NEXP * DFF * DMODEL);
    cvt(Wu, Wu_b, (long long)NEXP * DFF * DMODEL);
    cvt(Wd, Wd_b, (long long)NEXP * DMODEL * DFF);

    // attention path
    rmsnorm_kernel<<<TOKENS, 256, 0, stream>>>(x, anw, xn1_b, nullptr);
    gemm_bf16_kernel<<<dim3(DMODEL / 64, TOKENS / 128, 1), 256, 0, stream>>>(
        xn1_b, wq_b, qf, nullptr, nullptr, nullptr, nullptr, TOKENS, DMODEL, DMODEL, 0);
    gemm_bf16_kernel<<<dim3(KVDIM / 64, TOKENS / 128, 1), 256, 0, stream>>>(
        xn1_b, wk_b, kf, nullptr, nullptr, nullptr, nullptr, TOKENS, KVDIM, DMODEL, 0);
    gemm_bf16_kernel<<<dim3(KVDIM / 64, TOKENS / 128, 1), 256, 0, stream>>>(
        xn1_b, wv_b, vf, nullptr, nullptr, nullptr, nullptr, TOKENS, KVDIM, DMODEL, 0);
    rope_layout_kernel<<<TOKENS, 256, 0, stream>>>(qf, kf, vf, qh_b, kh_b, vt_b);
    flash_attn_kernel<<<dim3(SEQ / 16, NHEADS, BATCH), 32, 0, stream>>>(qh_b, kh_b, vt_b, attn_b);
    gemm_bf16_kernel<<<dim3(DMODEL / 64, TOKENS / 128, 1), 256, 0, stream>>>(
        attn_b, wo_b, h1, x, nullptr, nullptr, nullptr, TOKENS, DMODEL, DMODEL, 0);

    // MoE path
    rmsnorm_kernel<<<TOKENS, 256, 0, stream>>>(h1, fnw, xn2_b, xn2_f);
    router_kernel<<<TOKENS, 32, 0, stream>>>(xn2_f, rw, probs, tki, tkw);
    probsum_kernel<<<NEXP, 256, 0, stream>>>(probs, probsum);
    build_lists_kernel<<<1, 1, 0, stream>>>(tki, counts, offsets, rowlist, slotmap);
    moe_gu_kernel<<<dim3(DFF / 64, TOKENS / 128, NEXP), 256, 0, stream>>>(
        xn2_b, Wg_b, Wu_b, hbuf, rowlist, counts, offsets);
    gemm_bf16_kernel<<<dim3(DMODEL / 64, TOKENS / 128, NEXP), 256, 0, stream>>>(
        hbuf, Wd_b, ybuf, nullptr, nullptr, counts, offsets,
        TOKENS, DMODEL, DFF, (long long)DMODEL * DFF);
    combine_kernel<<<TOKENS, 256, 0, stream>>>(h1, ybuf, slotmap, tkw, out_x);
    finalize_kernel<<<1, 1, 0, stream>>>(counts, probsum, out_aux, out_tpe);
}
